// DecoderRNN_v2_2173253451848
// MI455X (gfx1250) — matrix operand order repaired and checked by reading, not compiled, not run on hardware
//
#include <hip/hip_runtime.h>
#include <hip/hip_bf16.h>
#include <math.h>

// ---------------- problem dims ----------------
#define B_   128
#define P_   196
#define E_   512
#define H_   512
#define V_   12000
#define T_   20
#define KCAT 1536    // [emb(512) | ctx(512) | h(512)]
#define G4H  2048    // 4*H

#define KSTEP 64     // K elements staged per LDS stage
#define LDSW  72     // 64 + 8 pad (bf16) -> 144B row stride, conflict-free frag loads

typedef __attribute__((ext_vector_type(16))) __bf16 v16bf;
typedef __attribute__((ext_vector_type(8)))  __bf16 v8bf;
typedef __attribute__((ext_vector_type(8)))  float  v8f;

#if __has_builtin(__builtin_amdgcn_tensor_load_to_lds) && __has_builtin(__builtin_amdgcn_s_wait_tensorcnt)
#define HAVE_TDM 1
typedef __attribute__((ext_vector_type(4))) unsigned int v4u_;
typedef __attribute__((ext_vector_type(8))) int          v8i_;
typedef __attribute__((ext_vector_type(4))) int          v4i_;
static __device__ __forceinline__ void tdm_load(v4u_ g0, v8i_ g1) {
    v4i_ z4 = {0, 0, 0, 0};
#if defined(__clang_major__) && (__clang_major__ >= 23)
    v8i_ z8 = {0, 0, 0, 0, 0, 0, 0, 0};
    __builtin_amdgcn_tensor_load_to_lds(g0, g1, z4, z4, z8, 0);
#else
    __builtin_amdgcn_tensor_load_to_lds(g0, g1, z4, z4, 0);
#endif
}
#else
#define HAVE_TDM 0
#endif

// ============================================================
// Block-cooperative bf16 NT GEMM: C[M,N] = A[M,K] * B[N,K]^T (+bias)
// Block = 256 threads (8 waves) computes a 128x64 C tile.
// Double-buffered LDS staging (KSTEP=64); per stage each wave does
// 2 k-substeps x 4 n-tiles = 8 wmma; A fragment reused 4x.
// EPI==0: C f32 = acc + bias[n]
// EPI==1: C bf16 = sigmoid(acc + bias[n]) * mul[m,n]   (beta-gate fusion)
// ============================================================
template<int EPI>
__global__ __launch_bounds__(256)
void gemm_bf16_nt(const __bf16* __restrict__ A, int lda,
                  const __bf16* __restrict__ Bw, int ldb,
                  const float* __restrict__ bias,
                  const float* __restrict__ mul, int ldmul,
                  void* __restrict__ Cout, int ldc, int K, int N)
{
    __shared__ __attribute__((aligned(16))) __bf16 sA[2][128][LDSW];
    __shared__ __attribute__((aligned(16))) __bf16 sB[2][64][LDSW];

    const int tid  = threadIdx.x;
    const int wv   = tid >> 5;
    const int lane = tid & 31;
    const int half = lane >> 4;
    const int r    = lane & 15;
    const int m0   = blockIdx.y * 128;
    const int n0   = blockIdx.x * 64;
    const int nstages = K / KSTEP;

    const int crow = tid >> 3;
    const int kc   = (tid & 7) * 8;
    const __bf16* gA0 = A + (size_t)(m0 + crow +  0) * lda + kc;
    const __bf16* gA1 = A + (size_t)(m0 + crow + 32) * lda + kc;
    const __bf16* gA2 = A + (size_t)(m0 + crow + 64) * lda + kc;
    const __bf16* gA3 = A + (size_t)(m0 + crow + 96) * lda + kc;
    int br0 = n0 + crow;      if (br0 >= N) br0 = N - 1;
    int br1 = n0 + crow + 32; if (br1 >= N) br1 = N - 1;
    const __bf16* gB0 = Bw + (size_t)br0 * ldb + kc;
    const __bf16* gB1 = Bw + (size_t)br1 * ldb + kc;

    __bf16* lA0 = &sA[0][crow +  0][kc];
    __bf16* lA1 = &sA[0][crow + 32][kc];
    __bf16* lA2 = &sA[0][crow + 64][kc];
    __bf16* lA3 = &sA[0][crow + 96][kc];
    __bf16* lB0 = &sB[0][crow +  0][kc];
    __bf16* lB1 = &sB[0][crow + 32][kc];
    const int bufA = 128 * LDSW;
    const int bufB =  64 * LDSW;

    union U { v8bf h[2]; v16bf v; };
    v8f acc0 = {}, acc1 = {}, acc2 = {}, acc3 = {};
    float4 ra0, ra1, ra2, ra3, rb0, rb1;

    ra0 = *(const float4*)(gA0); ra1 = *(const float4*)(gA1);
    ra2 = *(const float4*)(gA2); ra3 = *(const float4*)(gA3);
    rb0 = *(const float4*)(gB0); rb1 = *(const float4*)(gB1);

    for (int s = 0; s < nstages; ++s) {
        const int buf = s & 1;
        const int lo  = buf * bufA;
        const int loB = buf * bufB;
        *(float4*)(lA0 + lo)  = ra0;
        *(float4*)(lA1 + lo)  = ra1;
        *(float4*)(lA2 + lo)  = ra2;
        *(float4*)(lA3 + lo)  = ra3;
        *(float4*)(lB0 + loB) = rb0;
        *(float4*)(lB1 + loB) = rb1;
        __syncthreads();

        if (s + 1 < nstages) {
            const int kk = (s + 1) * KSTEP;
            ra0 = *(const float4*)(gA0 + kk); ra1 = *(const float4*)(gA1 + kk);
            ra2 = *(const float4*)(gA2 + kk); ra3 = *(const float4*)(gA3 + kk);
            rb0 = *(const float4*)(gB0 + kk); rb1 = *(const float4*)(gB1 + kk);
            if (kk + KSTEP < K) {
                __builtin_prefetch(gA0 + kk + KSTEP, 0, 3);
                __builtin_prefetch(gB0 + kk + KSTEP, 0, 3);
            }
        }

#pragma unroll
        for (int ks = 0; ks < 2; ++ks) {
            U a, b0, b1, b2, b3;
            const __bf16* ap = &sA[buf][wv * 16 + r][ks * 32 + half * 8];
            a.h[0]  = *(const v8bf*)(ap);
            a.h[1]  = *(const v8bf*)(ap + 16);
            const __bf16* bp = &sB[buf][r][ks * 32 + half * 8];
            b0.h[0] = *(const v8bf*)(bp);
            b0.h[1] = *(const v8bf*)(bp + 16);
            b1.h[0] = *(const v8bf*)(bp + 16 * LDSW);
            b1.h[1] = *(const v8bf*)(bp + 16 * LDSW + 16);
            b2.h[0] = *(const v8bf*)(bp + 32 * LDSW);
            b2.h[1] = *(const v8bf*)(bp + 32 * LDSW + 16);
            b3.h[0] = *(const v8bf*)(bp + 48 * LDSW);
            b3.h[1] = *(const v8bf*)(bp + 48 * LDSW + 16);
            acc0 = __builtin_amdgcn_wmma_f32_16x16x32_bf16(
                       false, a.v, false, b0.v, (short)0, acc0, false, false);
            acc1 = __builtin_amdgcn_wmma_f32_16x16x32_bf16(
                       false, a.v, false, b1.v, (short)0, acc1, false, false);
            acc2 = __builtin_amdgcn_wmma_f32_16x16x32_bf16(
                       false, a.v, false, b2.v, (short)0, acc2, false, false);
            acc3 = __builtin_amdgcn_wmma_f32_16x16x32_bf16(
                       false, a.v, false, b3.v, (short)0, acc3, false, false);
        }
    }

    const int mrow0 = m0 + wv * 16;
#define GEMM_EPILOG(ACC, NT)                                                   \
    do {                                                                       \
        const int n = n0 + (NT) * 16 + r;                                      \
        if (n < N) {                                                           \
            const float bn = bias ? bias[n] : 0.0f;                            \
            _Pragma("unroll")                                                  \
            for (int rr = 0; rr < 8; ++rr) {                                   \
                const int m = mrow0 + rr + half * 8;                           \
                float v = (ACC)[rr] + bn;                                      \
                if (EPI == 0) {                                                \
                    ((float*)Cout)[(size_t)m * ldc + n] = v;                   \
                } else {                                                       \
                    float sg = 1.0f / (1.0f + __expf(-v));                     \
                    ((__bf16*)Cout)[(size_t)m * ldc + n] =                     \
                        (__bf16)(sg * mul[(size_t)m * ldmul + n]);             \
                }                                                              \
            }                                                                  \
        }                                                                      \
    } while (0)
    GEMM_EPILOG(acc0, 0);
    GEMM_EPILOG(acc1, 1);
    GEMM_EPILOG(acc2, 2);
    GEMM_EPILOG(acc3, 3);
#undef GEMM_EPILOG
}

#if HAVE_TDM
// ============================================================
// TDM variant: B (weight) tiles DMA'd into LDS by the Tensor Data
// Mover; D# pad_enable inserts 4 DWORDs per 32-DWORD row so the LDS
// tile has the same 72-bf16 padded row stride as the manual path.
// A operand keeps register staging. Used for the logits GEMM.
// ============================================================
__global__ __launch_bounds__(256)
void gemm_bf16_tdm(const __bf16* __restrict__ A, int lda,
                   const __bf16* __restrict__ Bw, int ldb,
                   const float* __restrict__ bias,
                   float* __restrict__ Cout, int ldc, int K, int N)
{
    __shared__ __attribute__((aligned(16))) __bf16 sA[2][128][LDSW];
    __shared__ __attribute__((aligned(16))) __bf16 sB[2][64][LDSW];

    const int tid  = threadIdx.x;
    const int wv   = tid >> 5;
    const int lane = tid & 31;
    const int half = lane >> 4;
    const int r    = lane & 15;
    const int m0   = blockIdx.y * 128;
    const int n0   = blockIdx.x * 64;
    const int nstages = K / KSTEP;

    const int crow = tid >> 3;
    const int kc   = (tid & 7) * 8;
    const __bf16* gA0 = A + (size_t)(m0 + crow +  0) * lda + kc;
    const __bf16* gA1 = A + (size_t)(m0 + crow + 32) * lda + kc;
    const __bf16* gA2 = A + (size_t)(m0 + crow + 64) * lda + kc;
    const __bf16* gA3 = A + (size_t)(m0 + crow + 96) * lda + kc;
    __bf16* lA0 = &sA[0][crow +  0][kc];
    __bf16* lA1 = &sA[0][crow + 32][kc];
    __bf16* lA2 = &sA[0][crow + 64][kc];
    __bf16* lA3 = &sA[0][crow + 96][kc];
    const int bufA = 128 * LDSW;

    // ---- D# descriptor (ISA 8.3/8.4) ----
    // group1: data_size=1(2B) @16, pad_enable @20, pad_interval=4(32 DW) @22,
    //         pad_amount=3(4 DW) @25, tensor_dim0=K @48, tensor_dim1=N @80,
    //         tile_dim0=64 @112, tile_dim1=64 @128, dim0_stride=ldb @160
    const unsigned td0 = (unsigned)K, td1 = (unsigned)N;
    const unsigned long long st0 = (unsigned long long)ldb;
    const v8i_ g1v = {
        (int)((1u << 16) | (1u << 20) | (4u << 22) | (3u << 25)),
        (int)((td0 & 0xFFFFu) << 16),
        (int)(((td0 >> 16) & 0xFFFFu) | ((td1 & 0xFFFFu) << 16)),
        (int)(((td1 >> 16) & 0xFFFFu) | (64u << 16)),
        (int)(64u),
        (int)(st0 & 0xFFFFFFFFu),
        (int)((st0 >> 32) & 0xFFFFu),
        0 };
    const unsigned ldsB0 = (unsigned)(size_t)&sB[0][0][0];
    const unsigned ldsB1 = (unsigned)(size_t)&sB[1][0][0];
    const unsigned long long gbase =
        (unsigned long long)(size_t)(Bw + (size_t)n0 * ldb);

    union U { v8bf h[2]; v16bf v; };
    v8f acc0 = {}, acc1 = {}, acc2 = {}, acc3 = {};
    float4 ra0, ra1, ra2, ra3;

    ra0 = *(const float4*)(gA0); ra1 = *(const float4*)(gA1);
    ra2 = *(const float4*)(gA2); ra3 = *(const float4*)(gA3);

    if (wv == 0) {   // stage 0 tile via TDM (per-wave op, EXEC ignored)
        unsigned long long ga = gbase;
        v4u_ g0 = { 1u, ldsB0, (unsigned)(ga & 0xFFFFFFFFu),
                    (unsigned)(((ga >> 32) & 0x01FFFFFFu) | (2u << 30)) };
        tdm_load(g0, g1v);
    }

    for (int s = 0; s < nstages; ++s) {
        const int buf = s & 1;
        const int lo  = buf * bufA;
        *(float4*)(lA0 + lo) = ra0;
        *(float4*)(lA1 + lo) = ra1;
        *(float4*)(lA2 + lo) = ra2;
        *(float4*)(lA3 + lo) = ra3;
        __syncthreads();                    // prior readers of buf^1 are done

        if (wv == 0) {
            if (s + 1 < nstages) {          // issue next tile, then wait current
                unsigned long long ga = gbase + (unsigned long long)(s + 1) * KSTEP * 2u;
                v4u_ g0 = { 1u, buf ? ldsB0 : ldsB1, (unsigned)(ga & 0xFFFFFFFFu),
                            (unsigned)(((ga >> 32) & 0x01FFFFFFu) | (2u << 30)) };
                tdm_load(g0, g1v);
                __builtin_amdgcn_s_wait_tensorcnt(1);   // in-order: stage s done
            } else {
                __builtin_amdgcn_s_wait_tensorcnt(0);
            }
        }
        __syncthreads();                    // B tile of stage s visible to all

        if (s + 1 < nstages) {
            const int kk = (s + 1) * KSTEP;
            ra0 = *(const float4*)(gA0 + kk); ra1 = *(const float4*)(gA1 + kk);
            ra2 = *(const float4*)(gA2 + kk); ra3 = *(const float4*)(gA3 + kk);
        }

#pragma unroll
        for (int ks = 0; ks < 2; ++ks) {
            U a, b0, b1, b2, b3;
            const __bf16* ap = &sA[buf][wv * 16 + r][ks * 32 + half * 8];
            a.h[0]  = *(const v8bf*)(ap);
            a.h[1]  = *(const v8bf*)(ap + 16);
            const __bf16* bp = &sB[buf][r][ks * 32 + half * 8];
            b0.h[0] = *(const v8bf*)(bp);
            b0.h[1] = *(const v8bf*)(bp + 16);
            b1.h[0] = *(const v8bf*)(bp + 16 * LDSW);
            b1.h[1] = *(const v8bf*)(bp + 16 * LDSW + 16);
            b2.h[0] = *(const v8bf*)(bp + 32 * LDSW);
            b2.h[1] = *(const v8bf*)(bp + 32 * LDSW + 16);
            b3.h[0] = *(const v8bf*)(bp + 48 * LDSW);
            b3.h[1] = *(const v8bf*)(bp + 48 * LDSW + 16);
            acc0 = __builtin_amdgcn_wmma_f32_16x16x32_bf16(
                       false, a.v, false, b0.v, (short)0, acc0, false, false);
            acc1 = __builtin_amdgcn_wmma_f32_16x16x32_bf16(
                       false, a.v, false, b1.v, (short)0, acc1, false, false);
            acc2 = __builtin_amdgcn_wmma_f32_16x16x32_bf16(
                       false, a.v, false, b2.v, (short)0, acc2, false, false);
            acc3 = __builtin_amdgcn_wmma_f32_16x16x32_bf16(
                       false, a.v, false, b3.v, (short)0, acc3, false, false);
        }
    }

    const int mrow0 = m0 + wv * 16;
#define TDM_EPILOG(ACC, NT)                                                    \
    do {                                                                       \
        const int n = n0 + (NT) * 16 + r;                                      \
        if (n < N) {                                                           \
            const float bn = bias ? bias[n] : 0.0f;                            \
            _Pragma("unroll")                                                  \
            for (int rr = 0; rr < 8; ++rr) {                                   \
                const int m = mrow0 + rr + half * 8;                           \
                Cout[(size_t)m * ldc + n] = (ACC)[rr] + bn;                    \
            }                                                                  \
        }                                                                      \
    } while (0)
    TDM_EPILOG(acc0, 0);
    TDM_EPILOG(acc1, 1);
    TDM_EPILOG(acc2, 2);
    TDM_EPILOG(acc3, 3);
#undef TDM_EPILOG
}
#endif  // HAVE_TDM

// ---------------- setup / pack kernels ----------------
__global__ void cast_f32_bf16(const float* __restrict__ s, __bf16* __restrict__ d, int n) {
    int i = blockIdx.x * 256 + threadIdx.x;
    if (i < n) d[i] = (__bf16)s[i];
}

__global__ void pack_wcat(const float* __restrict__ Wih, const float* __restrict__ Whh,
                          __bf16* __restrict__ out) {
    int i = blockIdx.x * 256 + threadIdx.x;
    if (i >= G4H * KCAT) return;
    int k = i % KCAT, row = i / KCAT;
    float v = (k < 2 * E_) ? Wih[(size_t)row * (2 * E_) + k]
                           : Whh[(size_t)row * H_ + (k - 2 * E_)];
    out[i] = (__bf16)v;
}

__global__ void add_bias_vec(const float* __restrict__ a, const float* __restrict__ b,
                             float* __restrict__ o, int n) {
    int i = blockIdx.x * 256 + threadIdx.x;
    if (i < n) o[i] = a[i] + b[i];
}

__global__ void emb_gather(const float* __restrict__ table, const int* __restrict__ caps,
                           __bf16* __restrict__ out) {
    int i = blockIdx.x * 256 + threadIdx.x;
    if (i >= B_ * T_ * E_) return;
    int e = i & (E_ - 1);
    int bt = i >> 9;
    int cap = caps[bt];
    out[i] = (__bf16)table[(size_t)cap * E_ + e];
}

__global__ void init_state(float* __restrict__ h, float* __restrict__ c,
                           __bf16* __restrict__ xin) {
    int i = blockIdx.x * 256 + threadIdx.x;
    if (i >= B_ * H_) return;
    h[i] = 0.0f; c[i] = 0.0f;
    int b = i >> 9, j = i & (H_ - 1);
    xin[(size_t)b * KCAT + 2 * E_ + j] = (__bf16)0.0f;
}

// ---------------- attention: e -> softmax -> ctx ----------------
__global__ __launch_bounds__(256)
void attn_kernel(const float* __restrict__ att1,   // [B,P,E]
                 const float* __restrict__ att2,   // [B,E]
                 const float* __restrict__ wf,     // [E]
                 const float* __restrict__ bf_att, // [1]
                 const float* __restrict__ enc,    // [B,P,E]
                 const __bf16* __restrict__ embb,  // [B,T,E] bf16
                 __bf16* __restrict__ xin,         // [B,KCAT]
                 float* __restrict__ alpha_out,    // [B,T,P]
                 int t)
{
    const int b = blockIdx.x;
    const int tid = threadIdx.x;
    __shared__ float s_a2[E_];
    __shared__ float s_wf[E_];
    __shared__ float s_e[P_];
    __shared__ float s_red[256];

    for (int i = tid; i < E_; i += 256) { s_a2[i] = att2[b * E_ + i]; s_wf[i] = wf[i]; }
    __syncthreads();

    const int wv = tid >> 5, lane = tid & 31;
    const float* att1b = att1 + (size_t)b * P_ * E_;
    for (int p = wv; p < P_; p += 8) {
        const float* row = att1b + (size_t)p * E_;
        float loc = 0.0f;
        for (int a = lane; a < E_; a += 32) {
            float v = row[a] + s_a2[a];
            v = v > 0.0f ? v : 0.0f;          // relu
            loc += v * s_wf[a];
        }
        for (int off = 16; off > 0; off >>= 1) loc += __shfl_xor(loc, off, 32);
        if (lane == 0) s_e[p] = loc + bf_att[0];
    }
    __syncthreads();

    float ev = (tid < P_) ? s_e[tid] : -3.4e38f;
    s_red[tid] = ev; __syncthreads();
    for (int off = 128; off > 0; off >>= 1) {
        if (tid < off) s_red[tid] = fmaxf(s_red[tid], s_red[tid + off]);
        __syncthreads();
    }
    float mx = s_red[0]; __syncthreads();
    float ex = (tid < P_) ? __expf(ev - mx) : 0.0f;
    s_red[tid] = ex; __syncthreads();
    for (int off = 128; off > 0; off >>= 1) {
        if (tid < off) s_red[tid] += s_red[tid + off];
        __syncthreads();
    }
    float inv = 1.0f / s_red[0];
    __syncthreads();
    if (tid < P_) {
        float al = ex * inv;
        s_e[tid] = al;
        alpha_out[((size_t)b * T_ + t) * P_ + tid] = al;
    }
    __syncthreads();

    const float* encb = enc + (size_t)b * P_ * E_;
    for (int e = tid; e < E_; e += 256) {
        float accv = 0.0f;
        for (int p = 0; p < P_; ++p) accv += s_e[p] * encb[(size_t)p * E_ + e];
        xin[(size_t)b * KCAT + E_ + e] = (__bf16)accv;                        // ctx
        xin[(size_t)b * KCAT + e]      = embb[((size_t)b * T_ + t) * E_ + e]; // emb
    }
}

// ---------------- LSTM cell ----------------
__global__ void lstm_kernel(const float* __restrict__ gates,  // [B,4H] i,f,g,o
                            float* __restrict__ h, float* __restrict__ c,
                            __bf16* __restrict__ xin) {
    int i = blockIdx.x * 256 + threadIdx.x;
    if (i >= B_ * H_) return;
    int b = i >> 9, j = i & (H_ - 1);
    const float* g = gates + (size_t)b * G4H;
    float ii = g[j], ff = g[H_ + j], gg = g[2 * H_ + j], oo = g[3 * H_ + j];
    float si = 1.0f / (1.0f + __expf(-ii));
    float sf = 1.0f / (1.0f + __expf(-ff));
    float so = 1.0f / (1.0f + __expf(-oo));
    float cn = sf * c[i] + si * tanhf(gg);
    float hn = so * tanhf(cn);
    c[i] = cn; h[i] = hn;
    xin[(size_t)b * KCAT + 2 * E_ + j] = (__bf16)hn;
}

// ============================================================
extern "C" void kernel_launch(void* const* d_in, const int* in_sizes, int n_in,
                              void* d_out, int out_size, void* d_ws, size_t ws_size,
                              hipStream_t stream) {
    const float* enc       = (const float*)d_in[0];
    const int*   caps      = (const int*)  d_in[1];
    const float* emb_table = (const float*)d_in[2];
    const float* We        = (const float*)d_in[3];
    const float* be        = (const float*)d_in[4];
    const float* Wd        = (const float*)d_in[5];
    const float* bd        = (const float*)d_in[6];
    const float* Wf        = (const float*)d_in[7];
    const float* bfat      = (const float*)d_in[8];
    const float* Wih       = (const float*)d_in[9];
    const float* bih       = (const float*)d_in[10];
    const float* Whh       = (const float*)d_in[11];
    const float* bhh       = (const float*)d_in[12];
    const float* Wb        = (const float*)d_in[13];
    const float* bb        = (const float*)d_in[14];
    const float* Wfc       = (const float*)d_in[15];
    const float* bfc       = (const float*)d_in[16];

    float* outs   = (float*)d_out;                       // [B,T,V]
    float* alphas = outs + (size_t)B_ * T_ * V_;         // [B,T,P]

    char* wp = (char*)d_ws;
    auto alloc = [&](size_t bytes) {
        char* q = wp; wp += (bytes + 255) & ~(size_t)255; return q;
    };
    float*  att1  = (float*) alloc((size_t)B_ * P_ * E_ * 4);
    __bf16* encb  = (__bf16*)alloc((size_t)B_ * P_ * E_ * 2);
    __bf16* Web   = (__bf16*)alloc((size_t)E_ * E_ * 2);
    __bf16* Wdb   = (__bf16*)alloc((size_t)E_ * H_ * 2);
    __bf16* Wbb   = (__bf16*)alloc((size_t)E_ * H_ * 2);
    __bf16* Wfcb  = (__bf16*)alloc((size_t)V_ * H_ * 2);
    __bf16* Wcat  = (__bf16*)alloc((size_t)G4H * KCAT * 2);
    float*  bcat  = (float*) alloc((size_t)G4H * 4);
    __bf16* embb  = (__bf16*)alloc((size_t)B_ * T_ * E_ * 2);
    __bf16* xin   = (__bf16*)alloc((size_t)B_ * KCAT * 2);
    float*  att2  = (float*) alloc((size_t)B_ * E_ * 4);
    float*  gates = (float*) alloc((size_t)B_ * G4H * 4);
    float*  hbuf  = (float*) alloc((size_t)B_ * H_ * 4);
    float*  cbuf  = (float*) alloc((size_t)B_ * H_ * 4);
    __bf16* ybuf  = (__bf16*)alloc((size_t)B_ * H_ * 2);

    dim3 blk(256);
    auto g1 = [](size_t n) { return dim3((unsigned)((n + 255) / 256)); };
    auto gn = [](int n) { return (unsigned)((n + 63) / 64); };

    cast_f32_bf16<<<g1((size_t)B_*P_*E_), blk, 0, stream>>>(enc, encb, B_*P_*E_);
    cast_f32_bf16<<<g1((size_t)E_*E_),    blk, 0, stream>>>(We,  Web,  E_*E_);
    cast_f32_bf16<<<g1((size_t)E_*H_),    blk, 0, stream>>>(Wd,  Wdb,  E_*H_);
    cast_f32_bf16<<<g1((size_t)E_*H_),    blk, 0, stream>>>(Wb,  Wbb,  E_*H_);
    cast_f32_bf16<<<g1((size_t)V_*H_),    blk, 0, stream>>>(Wfc, Wfcb, V_*H_);
    pack_wcat    <<<g1((size_t)G4H*KCAT), blk, 0, stream>>>(Wih, Whh, Wcat);
    add_bias_vec <<<g1((size_t)G4H),      blk, 0, stream>>>(bih, bhh, bcat, G4H);
    emb_gather   <<<g1((size_t)B_*T_*E_), blk, 0, stream>>>(emb_table, caps, embb);
    init_state   <<<g1((size_t)B_*H_),    blk, 0, stream>>>(hbuf, cbuf, xin);

    // att1 = enc @ We^T + be
    gemm_bf16_nt<0><<<dim3(gn(E_), (B_*P_)/128), blk, 0, stream>>>(
        encb, E_, Web, E_, be, nullptr, 0, att1, E_, E_, E_);

    for (int t = 0; t < T_; ++t) {
        gemm_bf16_nt<0><<<dim3(gn(E_), 1), blk, 0, stream>>>(
            xin + 2*E_, KCAT, Wdb, H_, bd, nullptr, 0, att2, E_, H_, E_);

        attn_kernel<<<dim3(B_), blk, 0, stream>>>(
            att1, att2, Wf, bfat, enc, embb, xin, alphas, t);

        gemm_bf16_nt<0><<<dim3(gn(G4H), 1), blk, 0, stream>>>(
            xin, KCAT, Wcat, KCAT, bcat, nullptr, 0, gates, G4H, KCAT, G4H);

        lstm_kernel<<<g1((size_t)B_*H_), blk, 0, stream>>>(gates, hbuf, cbuf, xin);

        gemm_bf16_nt<1><<<dim3(gn(E_), 1), blk, 0, stream>>>(
            xin + 2*E_, KCAT, Wbb, H_, bb, hbuf, H_, ybuf, H_, H_, E_);

#if HAVE_TDM
        gemm_bf16_tdm<<<dim3(gn(V_), 1), blk, 0, stream>>>(
            ybuf, H_, Wfcb, H_, bfc, outs + (size_t)t * V_, T_ * V_, H_, V_);
#else
        gemm_bf16_nt<0><<<dim3(gn(V_), 1), blk, 0, stream>>>(
            ybuf, H_, Wfcb, H_, bfc, nullptr, 0,
            outs + (size_t)t * V_, T_ * V_, H_, V_);
#endif
    }
}
